// AutoCorrelationAttention_52561809769178
// MI455X (gfx1250) — matrix operand order repaired and checked by reading, not compiled, not run on hardware
//
#include <hip/hip_runtime.h>
#include <hip/hip_bf16.h>

#define B_   16
#define L_   2048
#define DM_  1024
#define H_   16
#define DK_  64
#define TOPK_ 7

typedef __attribute__((ext_vector_type(16))) _Float16 v16h;
typedef __attribute__((ext_vector_type(8)))  float    v8f;
typedef __attribute__((ext_vector_type(4)))  int      v4i;
typedef __attribute__((ext_vector_type(4)))  unsigned int u32x4;
typedef __attribute__((ext_vector_type(8)))  int      i32x8;
typedef __attribute__((ext_vector_type(4)))  int      i32x4;

// ---------------------------------------------------------------------------
// CDNA5 async global->LDS path (ASYNCcnt).
// Builtin signature (verified round 2): (global v4i*, lds v4i*, imm, imm).
// ---------------------------------------------------------------------------
#if defined(__has_builtin)
#if __has_builtin(__builtin_amdgcn_global_load_async_to_lds_b128)
#define HAVE_ASYNC_LDS 1
#endif
#if __has_builtin(__builtin_amdgcn_tensor_load_to_lds)
#define HAVE_TDM 1
#endif
#endif
#ifndef HAVE_ASYNC_LDS
#define HAVE_ASYNC_LDS 0
#endif
#ifndef HAVE_TDM
#define HAVE_TDM 0
#endif

#if HAVE_ASYNC_LDS
typedef __attribute__((address_space(1))) v4i gv4i;  // global int4
typedef __attribute__((address_space(3))) v4i lv4i;  // LDS int4
#define ASYNC_CP16(gsrc, ldst) \
  __builtin_amdgcn_global_load_async_to_lds_b128((gv4i*)(gsrc), (lv4i*)(ldst), 0, 0)
__device__ __forceinline__ void wait_async0() {
#if __has_builtin(__builtin_amdgcn_s_wait_asynccnt)
  __builtin_amdgcn_s_wait_asynccnt(0);
#else
  asm volatile("s_wait_asynccnt 0x0" ::: "memory");
#endif
}
#endif

#if HAVE_TDM
__device__ __forceinline__ void wait_tensor0() {
#if __has_builtin(__builtin_amdgcn_s_wait_tensorcnt)
  __builtin_amdgcn_s_wait_tensorcnt(0);
#else
  asm volatile("s_wait_tensorcnt 0x0" ::: "memory");
#endif
}

// TDM bulk copy: 2048 contiguous f32 (one 1x2048 tile of a 2048x1 tensor)
// global -> LDS.  D# per ISA sec 8.3/8.4.  Issue once per workgroup.
__device__ __forceinline__ void tdm_load_series(const float* gsrc, float* ldst) {
  const unsigned lds = (unsigned)(size_t)(__attribute__((address_space(3))) char*)ldst;
  const unsigned long long ga = (unsigned long long)(size_t)gsrc;
  u32x4 g0;
  g0[0] = 1u;                                                 // count=1 (valid)
  g0[1] = lds;                                                // lds_addr
  g0[2] = (unsigned)(ga & 0xffffffffu);                       // global_addr[31:0]
  g0[3] = (unsigned)((ga >> 32) & 0x01ffffffu) | (2u << 30);  // addr[56:32]|type=2
  i32x8 g1;
  g1[0] = (int)(2u << 16);        // workgroup_mask=0, data_size=2 (4 bytes)
  g1[1] = (int)(2048u << 16);     // tensor_dim0[15:0]=2048 in [63:48]
  g1[2] = (int)(1u << 16);        // tensor_dim0 hi=0, tensor_dim1[15:0]=1
  g1[3] = (int)(2048u << 16);     // tile_dim0=2048 in [127:112]
  g1[4] = 1;                      // tile_dim1=1, tile_dim2=0
  g1[5] = 2048;                   // tensor_dim0_stride[31:0]
  g1[6] = (int)(2048u << 16);     // stride0 hi=0, tensor_dim1_stride[15:0]
  g1[7] = 0;
  i32x4 gz = {0, 0, 0, 0};        // groups 2/3 unused (<=2D tensor)
#if __clang_major__ >= 23
  i32x8 gz8 = {0, 0, 0, 0, 0, 0, 0, 0};
  __builtin_amdgcn_tensor_load_to_lds(g0, g1, gz, gz, gz8, 0);
#else
  __builtin_amdgcn_tensor_load_to_lds(g0, g1, gz, gz, 0);
#endif
}
#endif  // HAVE_TDM

// ---------------------------------------------------------------------------
// WMMA helpers (CDNA5: V_WMMA_F32_16X16X32_F16, f32 accumulate)
// ---------------------------------------------------------------------------
__device__ __forceinline__ v8f wmma_f16(v16h a, v16h b, v8f c) {
  return __builtin_amdgcn_wmma_f32_16x16x32_f16(
      /*neg_a=*/false, a, /*neg_b=*/false, b,
      /*c_mod=*/(short)0, c, /*reuse_a=*/false, /*reuse_b=*/false);
}

// A fragment (16x32, f16) from row-major f32 matrix X[ld], rows row0..row0+15.
// Per ISA 7.12.2: lanes 0-15 hold row M=lane, K=0..7 & 16..23; lanes 16-31
// hold row M=lane-16, K=8..15 & 24..31.
__device__ __forceinline__ v16h load_A_f32(const float* __restrict__ X, int ld,
                                           int row0, int k0, int lane) {
  const int r = row0 + (lane & 15);
  const float* p = X + (size_t)r * ld + k0 + ((lane >> 4) << 3);
  v16h a;
#pragma unroll
  for (int i = 0; i < 8; ++i) a[i] = (_Float16)p[i];
#pragma unroll
  for (int i = 0; i < 8; ++i) a[8 + i] = (_Float16)p[16 + i];
  return a;
}

// B fragment (32x16, f16) where B[k,n] = W16[n0+n, k0+k]; weights are
// pre-converted to f16 so this is two 16B vector loads per lane.
__device__ __forceinline__ v16h load_B_f16(const _Float16* __restrict__ W, int ld,
                                           int n0, int k0, int lane) {
  const _Float16* p = W + (size_t)(n0 + (lane & 15)) * ld + k0 + ((lane >> 4) << 3);
  typedef __attribute__((ext_vector_type(8))) _Float16 v8h_;
  const v8h_ lo = *(const v8h_*)p;          // K = 8h..8h+7, 16B aligned
  const v8h_ hi = *(const v8h_*)(p + 16);   // K = 16+8h..16+8h+7
  return __builtin_shufflevector(lo, hi, 0,1,2,3,4,5,6,7,8,9,10,11,12,13,14,15);
}

// ---------------------------------------------------------------------------
// Kernel 0: one-time f32 -> f16 conversion of a weight matrix (L2-resident).
// ---------------------------------------------------------------------------
__global__ __launch_bounds__(256) void cvt_f16_kernel(const float* __restrict__ src,
                                                      _Float16* __restrict__ dst,
                                                      int n) {
  const int i = blockIdx.x * 256 + threadIdx.x;
  if (i < n) dst[i] = (_Float16)src[i];
}

// ---------------------------------------------------------------------------
// Kernel 1: projection GEMM  Y[b,h,dk,l] = sum_k X[b*L+l, k] * W[h*DK+dk, k]
// One wave computes a 16(l) x 64(n) tile: A fragment reused across 4 B tiles.
// ---------------------------------------------------------------------------
__global__ __launch_bounds__(32) void proj_kernel(const float* __restrict__ X,
                                                  const _Float16* __restrict__ W16,
                                                  float* __restrict__ Y) {
  const int lane = threadIdx.x;
  const int row0 = blockIdx.x * 16;   // flat (b*L + l) tile base
  const int n0   = blockIdx.y * 64;   // output-feature tile base
  v8f acc[4] = {};
  for (int k0 = 0; k0 < DM_; k0 += 32) {
    if (k0 + 32 < DM_)  // prefetch next A K-slab, near locality (WGP scope)
      __builtin_prefetch(X + (size_t)(row0 + (lane & 15)) * DM_ + k0 + 32, 0, 3);
    const v16h a = load_A_f32(X, DM_, row0, k0, lane);
#pragma unroll
    for (int s = 0; s < 4; ++s)
      acc[s] = wmma_f16(a, load_B_f16(W16, DM_, n0 + 16 * s, k0, lane), acc[s]);
  }
  // D layout: lane has fixed N = lane&15, M = r + (lane>=16 ? 8 : 0) per VGPR r
  // -> 8 consecutive l values per lane: contiguous store in [B,H,DK,L].
  const int b  = row0 / L_;
  const int l0 = (row0 % L_) + ((lane >> 4) << 3);
#pragma unroll
  for (int s = 0; s < 4; ++s) {
    const int n = n0 + 16 * s + (lane & 15);
    const int h = n >> 6, dk = n & 63;
    float* yp = Y + (((size_t)b * H_ + h) * DK_ + dk) * L_ + l0;
#pragma unroll
    for (int r = 0; r < 8; ++r) yp[r] = acc[s][r];
  }
}

// ---------------------------------------------------------------------------
// Kernel 2: per-series circular cross-correlation via LDS Stockham FFT.
// corr = Re( IFFT( FFT(Q) * conj(FFT(K)) ) ) / L     (length-2048 complex FFT)
// Series staged into LDS by the Tensor Data Mover (TENSORcnt) when available,
// else by async global->LDS b128 copies (ASYNCcnt).
// ---------------------------------------------------------------------------
__device__ float2* fft_stages(float2* buf0, float2* buf1, int tid, float sgn) {
  float2* src = buf0;
  float2* dst = buf1;
  const float base = sgn * 6.283185307179586f / (float)L_;
  for (int sh = 0; sh < 11; ++sh) {            // 11 radix-2 stages
    const int s = 1 << sh;
    for (int i = tid; i < L_ / 2; i += 256) {  // 1024 butterflies, 4/thread
      const int p = i >> sh;
      const int q = i & (s - 1);
      const float2 x0 = src[q + s * p];
      const float2 x1 = src[q + s * p + (L_ / 2)];  // s*(p+m), s*m == L/2
      const float ang = base * (float)(p << sh);    // sgn*2pi*p*s/L
      float sn, cs;
      __sincosf(ang, &sn, &cs);
      const float dx = x0.x - x1.x, dy = x0.y - x1.y;
      dst[q + 2 * s * p]     = make_float2(x0.x + x1.x, x0.y + x1.y);
      dst[q + 2 * s * p + s] = make_float2(dx * cs - dy * sn, dx * sn + dy * cs);
    }
    __syncthreads();
    float2* t = src; src = dst; dst = t;
  }
  return src;  // buffer holding the final result
}

// Load 2048 f32 series values into bufA as complex, staging raw reals through
// `scratch` in LDS.
__device__ __forceinline__ void load_series(const float* __restrict__ src,
                                            float2* bufA, float* scratch, int tid) {
#if HAVE_TDM
  if (tid == 0) {               // one TDM descriptor per workgroup (EXEC ignored)
    tdm_load_series(src, scratch);
    wait_tensor0();
  }
  __syncthreads();
  for (int i = tid; i < L_; i += 256) bufA[i] = make_float2(scratch[i], 0.f);
#elif HAVE_ASYNC_LDS
  for (int c = tid; c < L_ / 4; c += 256)
    ASYNC_CP16(src + c * 4, scratch + c * 4);
  wait_async0();
  __syncthreads();
  for (int i = tid; i < L_; i += 256) bufA[i] = make_float2(scratch[i], 0.f);
#else
  (void)scratch;
  for (int i = tid; i < L_; i += 256) bufA[i] = make_float2(src[i], 0.f);
#endif
  __syncthreads();
}

__global__ __launch_bounds__(256) void fft_corr_kernel(const float* __restrict__ Qp,
                                                       const float* __restrict__ Kp,
                                                       float* __restrict__ corr) {
  __shared__ __align__(16) float2 bufA[L_];   // 16 KB
  __shared__ __align__(16) float2 bufB[L_];   // 16 KB
  __shared__ __align__(16) float2 bufQ[L_];   // 16 KB (48 KB total < 320 KB/WGP)
  const int tid = threadIdx.x;
  const size_t base = (size_t)blockIdx.x * L_;  // series (b,h,dk)

  load_series(Qp + base, bufA, (float*)bufB, tid);
  float2* resQ = fft_stages(bufA, bufB, tid, -1.f);
  for (int i = tid; i < L_; i += 256) bufQ[i] = resQ[i];
  __syncthreads();

  load_series(Kp + base, bufA, (float*)bufB, tid);
  float2* resK = fft_stages(bufA, bufB, tid, -1.f);

  float2* prod = (resK == bufA) ? bufB : bufA;
  for (int i = tid; i < L_; i += 256) {
    const float2 qf = bufQ[i], kf = resK[i];
    prod[i] = make_float2(qf.x * kf.x + qf.y * kf.y,   // qf * conj(kf)
                          qf.y * kf.x - qf.x * kf.y);
  }
  __syncthreads();

  float2* resI = fft_stages(prod, resK, tid, 1.f);
  const float inv = 1.f / (float)L_;
  for (int i = tid; i < L_; i += 256) corr[base + i] = resI[i].x * inv;
}

// ---------------------------------------------------------------------------
// Kernel 3: mean over heads  mean[b,dk,l] = (1/H) * sum_h corr[b,h,dk,l]
// ---------------------------------------------------------------------------
__global__ __launch_bounds__(256) void mean_kernel(const float* __restrict__ corr,
                                                   float* __restrict__ meanB,
                                                   int total) {
  const int idx = blockIdx.x * 256 + threadIdx.x;
  if (idx >= total) return;
  const int l  = idx & (L_ - 1);
  const int t  = idx / L_;
  const int dk = t & (DK_ - 1);
  const int b  = t / DK_;
  float s = 0.f;
#pragma unroll
  for (int h = 0; h < H_; ++h)
    s += corr[(((size_t)b * H_ + h) * DK_ + dk) * L_ + l];
  meanB[idx] = s * (1.f / (float)H_);
}

// ---------------------------------------------------------------------------
// Kernel 4: top-7 along L per (b,dk) row + softmax over the 7 weights.
// ---------------------------------------------------------------------------
__global__ __launch_bounds__(256) void topk_kernel(const float* __restrict__ meanB,
                                                   float* __restrict__ wbuf,
                                                   int* __restrict__ dbuf) {
  __shared__ float sval[256];
  __shared__ int   sidx[256];
  __shared__ int   picked[TOPK_];
  __shared__ float pval[TOPK_];
  const int tid = threadIdx.x;
  const float* m = meanB + (size_t)blockIdx.x * L_;

  for (int it = 0; it < TOPK_; ++it) {
    float best = -3.402823466e38f;
    int bidx = 0;
    for (int l = tid; l < L_; l += 256) {
      bool skip = false;
      for (int j = 0; j < it; ++j) skip = skip || (picked[j] == l);
      const float v = m[l];
      if (!skip && v > best) { best = v; bidx = l; }
    }
    sval[tid] = best; sidx[tid] = bidx;
    __syncthreads();
    for (int off = 128; off > 0; off >>= 1) {
      if (tid < off && sval[tid + off] > sval[tid]) {
        sval[tid] = sval[tid + off];
        sidx[tid] = sidx[tid + off];
      }
      __syncthreads();
    }
    if (tid == 0) { picked[it] = sidx[0]; pval[it] = sval[0]; }
    __syncthreads();
  }
  if (tid == 0) {
    float mx = pval[0];
    for (int j = 1; j < TOPK_; ++j) mx = fmaxf(mx, pval[j]);
    float e[TOPK_], se = 0.f;
    for (int j = 0; j < TOPK_; ++j) { e[j] = __expf(pval[j] - mx); se += e[j]; }
    const float inv = 1.f / se;
    for (int j = 0; j < TOPK_; ++j) {
      wbuf[(size_t)blockIdx.x * 8 + j] = e[j] * inv;
      dbuf[(size_t)blockIdx.x * 8 + j] = picked[j];
    }
  }
}

// ---------------------------------------------------------------------------
// Kernel 5: delay aggregation (broadcast quirk: weights indexed by HEAD).
// agg stored [B, DM, L] (coalesced; output GEMM transposes through LDS).
// ---------------------------------------------------------------------------
__global__ __launch_bounds__(256) void agg_kernel(const float* __restrict__ Vp,
                                                  const float* __restrict__ wbuf,
                                                  const int* __restrict__ dbuf,
                                                  float* __restrict__ agg) {
  const int sid = blockIdx.x;              // b*H*DK + h*DK + dk  == b*DM + n
  const int dk = sid & (DK_ - 1);
  const int h  = (sid / DK_) & (H_ - 1);
  const int row = h * DK_ + dk;            // weights row keyed by (h, dk)
  __shared__ float w[TOPK_];
  __shared__ int   dly[TOPK_];
  if (threadIdx.x < TOPK_) {
    w[threadIdx.x]   = wbuf[(size_t)row * 8 + threadIdx.x];
    dly[threadIdx.x] = dbuf[(size_t)row * 8 + threadIdx.x];
  }
  __syncthreads();
  const float* v = Vp + (size_t)sid * L_;
  float* ag = agg + (size_t)sid * L_;
  for (int l = threadIdx.x; l < L_; l += 256) {
    float s = 0.f;
#pragma unroll
    for (int i = 0; i < TOPK_; ++i) {
      int idx = l + dly[i];
      if (idx >= L_) idx -= L_;
      s += v[idx] * w[i];
    }
    ag[l] = s;
  }
}

// ---------------------------------------------------------------------------
// Kernel 6: output GEMM  Out[b,l,n] = sum_k agg[b,k,l] * Wo[n,k] + bo[n].
// A operand is [B,DM,L] (k-major): a 32k x 16m f32 tile is staged to LDS via
// async global->LDS b128 copies, then the WMMA A fragment is built from ds
// loads with in-register f16 conversion.
// ---------------------------------------------------------------------------
__global__ __launch_bounds__(32) void out_gemm_kernel(const float* __restrict__ Agg,
                                                      const _Float16* __restrict__ Wo16,
                                                      const float* __restrict__ bo,
                                                      float* __restrict__ Out) {
  __shared__ __align__(16) float Atf[32][16];   // 2 KB staging tile
  const int lane = threadIdx.x;
  const int row0 = blockIdx.x * 16;
  const int n0   = blockIdx.y * 64;
  const int b  = row0 / L_;
  const int l0 = row0 % L_;
  v8f acc[4] = {};
  for (int k0 = 0; k0 < DM_; k0 += 32) {
#if HAVE_ASYNC_LDS
    // 32 rows x 64B: 128 b128 transfers, 4 per lane, all 16B aligned.
#pragma unroll
    for (int t = 0; t < 4; ++t) {
      const int tt  = t * 32 + lane;
      const int kr  = tt >> 2, seg = tt & 3;
      ASYNC_CP16(&Agg[((size_t)b * DM_ + k0 + kr) * L_ + l0 + seg * 4],
                 &Atf[kr][seg * 4]);
    }
    wait_async0();
#else
#pragma unroll
    for (int it = 0; it < 16; ++it) {
      const int kr = it * 2 + (lane >> 4);
      const int mm = lane & 15;
      Atf[kr][mm] = Agg[((size_t)b * DM_ + k0 + kr) * L_ + l0 + mm];
    }
#endif
    __syncthreads();
    v16h a;
    const int r  = lane & 15;
    const int kb = (lane >> 4) << 3;
#pragma unroll
    for (int i = 0; i < 8; ++i) a[i]     = (_Float16)Atf[kb + i][r];
#pragma unroll
    for (int i = 0; i < 8; ++i) a[8 + i] = (_Float16)Atf[16 + kb + i][r];
#pragma unroll
    for (int s = 0; s < 4; ++s)
      acc[s] = wmma_f16(a, load_B_f16(Wo16, DM_, n0 + 16 * s, k0, lane), acc[s]);
    __syncthreads();
  }
  const int lrow = l0 + ((lane >> 4) << 3);
#pragma unroll
  for (int s = 0; s < 4; ++s) {
    const int n = n0 + 16 * s + (lane & 15);
    const float bias = bo[n];
#pragma unroll
    for (int r = 0; r < 8; ++r)
      Out[((size_t)b * L_ + lrow + r) * DM_ + n] = acc[s][r] + bias;
  }
}

// ---------------------------------------------------------------------------
// Host launcher
// ---------------------------------------------------------------------------
extern "C" void kernel_launch(void* const* d_in, const int* in_sizes, int n_in,
                              void* d_out, int out_size, void* d_ws, size_t ws_size,
                              hipStream_t stream) {
  (void)in_sizes; (void)n_in; (void)out_size; (void)ws_size;
  const float* queries = (const float*)d_in[0];
  const float* keys    = (const float*)d_in[1];
  const float* values  = (const float*)d_in[2];
  const float* Wq = (const float*)d_in[3];
  const float* Wk = (const float*)d_in[4];
  const float* Wv = (const float*)d_in[5];
  const float* Wo = (const float*)d_in[6];
  const float* bo = (const float*)d_in[7];

  float* outP  = (float*)d_out;                       // [B,L,DM]
  float* corrP = outP + (size_t)B_ * L_ * DM_;        // [B,H,DK,L]

  float* ws = (float*)d_ws;
  const size_t NS = (size_t)B_ * H_ * DK_ * L_;       // 33.5M floats per tensor
  float* Qp    = ws;                                  // [B,H,DK,L]
  float* Kp    = ws + NS;
  float* Vp    = ws + 2 * NS;
  float* meanB = ws + 3 * NS;                         // [B,DK,L]
  float* wbuf  = meanB + (size_t)B_ * DK_ * L_;       // [B*DK, 8]
  int*   dbuf  = (int*)(wbuf + (size_t)B_ * DK_ * 8); // [B*DK, 8]
  _Float16* Wq16 = (_Float16*)(dbuf + (size_t)B_ * DK_ * 8);
  _Float16* Wk16 = Wq16 + (size_t)DM_ * DM_;
  _Float16* Wv16 = Wk16 + (size_t)DM_ * DM_;
  _Float16* Wo16 = Wv16 + (size_t)DM_ * DM_;
  float* agg = Qp;  // Qp dead after fft_corr_kernel -> reuse for agg [B,DM,L]

  const int nW = DM_ * DM_;
  const dim3 gcvt((nW + 255) / 256);
  cvt_f16_kernel<<<gcvt, 256, 0, stream>>>(Wq, Wq16, nW);
  cvt_f16_kernel<<<gcvt, 256, 0, stream>>>(Wk, Wk16, nW);
  cvt_f16_kernel<<<gcvt, 256, 0, stream>>>(Wv, Wv16, nW);
  cvt_f16_kernel<<<gcvt, 256, 0, stream>>>(Wo, Wo16, nW);

  const dim3 ggemm((B_ * L_) / 16, DM_ / 64);
  proj_kernel<<<ggemm, 32, 0, stream>>>(queries, Wq16, Qp);
  proj_kernel<<<ggemm, 32, 0, stream>>>(keys,    Wk16, Kp);
  proj_kernel<<<ggemm, 32, 0, stream>>>(values,  Wv16, Vp);

  fft_corr_kernel<<<B_ * H_ * DK_, 256, 0, stream>>>(Qp, Kp, corrP);

  const int meanN = B_ * DK_ * L_;
  mean_kernel<<<(meanN + 255) / 256, 256, 0, stream>>>(corrP, meanB, meanN);

  topk_kernel<<<B_ * DK_, 256, 0, stream>>>(meanB, wbuf, dbuf);

  agg_kernel<<<B_ * H_ * DK_, 256, 0, stream>>>(Vp, wbuf, dbuf, agg);

  out_gemm_kernel<<<ggemm, 32, 0, stream>>>(agg, Wo16, bo, outP);
}
